// RPN_80771154969282
// MI455X (gfx1250) — matrix operand order repaired and checked by reading, not compiled, not run on hardware
//
#include <hip/hip_runtime.h>
#include <hip/hip_bf16.h>

// ---------------------------------------------------------------------------
// RPN forward for MI455X (gfx1250): implicit-GEMM 3x3 conv via WMMA f16,
// VALU heads + box decode, bisection top-k, greedy NMS.
// ---------------------------------------------------------------------------

typedef __attribute__((ext_vector_type(16))) _Float16 v16h;
typedef __attribute__((ext_vector_type(8)))  float    v8f;

#define FHh 100
#define FWw 160
#define CIN 512
#define NPIX (FHh * FWw)          // 16000
#define NANCH (NPIX * 9)          // 144000
#define PADH 102
#define PADW 162
#define PRE_NMS 6000
#define POST_NMS 300
#define NEGV (-1e9f)

// ---------------- prep: pad + convert feature map to f16 -------------------
__global__ void pad_feat_f16(const float* __restrict__ feat, _Float16* __restrict__ fp) {
    int idx = blockIdx.x * blockDim.x + threadIdx.x;           // [PADH*PADW*CIN]
    int c  = idx & (CIN - 1);
    int p  = idx >> 9;
    int xx = p % PADW;
    int yy = p / PADW;
    float v = 0.f;
    if (yy >= 1 && yy <= FHh && xx >= 1 && xx <= FWw)
        v = feat[((size_t)(yy - 1) * FWw + (xx - 1)) * CIN + c];
    fp[idx] = (_Float16)v;
}

// ---------------- prep: pack W1 into B-fragment layout ---------------------
// layout: [tap(9)][kc(16)][octile(32)][lane(32)][e(16)]
// lane n (0..15): column oc = octile*16+n, K = kc*32 + (e&7) + (e>>3)*16      (e = 0..15)
// lane n (16..31): column oc = octile*16+(n-16), K = kc*32 + 8 + (e&7) + (e>>3)*16
__global__ void pack_w1_f16(const float* __restrict__ W1, _Float16* __restrict__ wp) {
    int idx = blockIdx.x * blockDim.x + threadIdx.x;           // 9*16*32*32*16
    int e = idx & 15;
    int r = idx >> 4;
    int lane = r & 31;  r >>= 5;
    int octile = r & 31; r >>= 5;
    int kc = r & 15;
    int tap = r >> 4;
    int oc = octile * 16 + (lane & 15);
    int ic = kc * 32 + (lane >> 4) * 8 + (e & 7) + (e >> 3) * 16;
    float v = W1[((size_t)tap * CIN + ic) * CIN + oc];
    wp[idx] = (_Float16)v;
}

// ---------------- main conv: implicit GEMM with WMMA -----------------------
union H8x2 { uint4 u[2]; v16h v; };

__device__ inline v16h load_a_frag(const _Float16* p) {
    // elements 0..7 at p, elements 8..15 at p+16 halves (A 16x32 f16 layout)
    H8x2 t;
    t.u[0] = *(const uint4*)(p);
    t.u[1] = *(const uint4*)(p + 16);
    return t.v;
}

__global__ __launch_bounds__(256) void conv3x3_wmma(
    const _Float16* __restrict__ featpad,   // [PADH][PADW][CIN]
    const _Float16* __restrict__ wpack,     // packed B fragments
    const float*    __restrict__ b1,
    float*          __restrict__ y)         // [NPIX][CIN]
{
    const int lane = threadIdx.x & 31;
    const int wave = threadIdx.x >> 5;      // 0..7 -> oc group of 64
    const int pixbase = blockIdx.x * 32;    // 32 pixels, same feature row
    const int fy  = pixbase / FWw;
    const int fx0 = pixbase % FWw;
    const int ocbase = wave * 64;

    const int m  = lane & 15;               // A-matrix row within tile
    const int kb = (lane >> 4) * 8;         // A K-offset: lanes 0-15 -> 0, 16-31 -> 8

    v8f acc[2][4];
#pragma unroll
    for (int mt = 0; mt < 2; ++mt)
#pragma unroll
        for (int ot = 0; ot < 4; ++ot)
#pragma unroll
            for (int r = 0; r < 8; ++r) acc[mt][ot][r] = 0.f;

    for (int dy = 0; dy < 3; ++dy) {
        for (int dx = 0; dx < 3; ++dx) {
            const int tap = dy * 3 + dx;
            const _Float16* arow0 =
                featpad + ((size_t)(fy + dy) * PADW + (fx0 + m + dx)) * CIN;
            const _Float16* arow1 = arow0 + (size_t)16 * CIN;   // second M-tile (+16 px)
            for (int kc = 0; kc < 16; ++kc) {
                const int ic0 = kc * 32 + kb;
                v16h a0 = load_a_frag(arow0 + ic0);
                v16h a1 = load_a_frag(arow1 + ic0);
                const _Float16* wkc =
                    wpack + ((((size_t)tap * 16 + kc) * 32 + (ocbase >> 4)) * 32 + lane) * 16;
#pragma unroll
                for (int ot = 0; ot < 4; ++ot) {
                    v16h b = *(const v16h*)(wkc + (size_t)ot * 512);
                    acc[0][ot] = __builtin_amdgcn_wmma_f32_16x16x32_f16(
                        false, a0, false, b, (short)0, acc[0][ot], false, false);
                    acc[1][ot] = __builtin_amdgcn_wmma_f32_16x16x32_f16(
                        false, a1, false, b, (short)0, acc[1][ot], false, false);
                }
            }
        }
    }

    // epilogue: bias + relu, scatter per documented C layout
    const int nn = lane & 15;
    const int mhalf = (lane >> 4) * 8;
#pragma unroll
    for (int ot = 0; ot < 4; ++ot) {
        const int oc = ocbase + ot * 16 + nn;
        const float bias = b1[oc];
#pragma unroll
        for (int mt = 0; mt < 2; ++mt) {
#pragma unroll
            for (int r = 0; r < 8; ++r) {
                float v = acc[mt][ot][r] + bias;
                v = v > 0.f ? v : 0.f;
                const int pix = pixbase + mt * 16 + mhalf + r;
                y[(size_t)pix * CIN + oc] = v;
            }
        }
    }
}

// ---------------- heads (1x1 convs) + box decode ---------------------------
__global__ __launch_bounds__(64) void heads_decode(
    const float* __restrict__ y,
    const float* __restrict__ Wc, const float* __restrict__ bc,
    const float* __restrict__ Wr, const float* __restrict__ br,
    const float* __restrict__ anchor_map,
    float* __restrict__ out_scores,   // [NPIX*9]
    float* __restrict__ out_reg,      // [NPIX*36]
    float* __restrict__ props,        // [NANCH][4]
    float* __restrict__ obj,          // raw sigmoid scores [NANCH]
    float* __restrict__ scmask)       // min-size-masked scores [NANCH]
{
    __shared__ float ylds[CIN];
    __shared__ float rlds[36];
    __shared__ float slds[9];
    const int pix = blockIdx.x;
    const int t = threadIdx.x;
    for (int i = t; i < CIN; i += 64) ylds[i] = y[(size_t)pix * CIN + i];
    __syncthreads();

    if (t < 45) {
        const bool is_s = (t < 9);
        const float* wb = is_s ? (Wc + t) : (Wr + (t - 9));
        const int stride = is_s ? 9 : 36;
        float acc = 0.f;
        for (int ic = 0; ic < CIN; ++ic) acc += ylds[ic] * wb[(size_t)ic * stride];
        if (is_s) {
            float s = 1.f / (1.f + __expf(-(acc + bc[t])));
            out_scores[(size_t)pix * 9 + t] = s;
            slds[t] = s;
        } else {
            float rv = acc + br[t - 9];
            out_reg[(size_t)pix * 36 + (t - 9)] = rv;
            rlds[t - 9] = rv;
        }
    }
    __syncthreads();

    if (t < 9) {
        const float* an = anchor_map + (size_t)pix * 36 + t * 4;
        float ay1 = an[0], ax1 = an[1], ay2 = an[2], ax2 = an[3];
        float ah = ay2 - ay1, aw = ax2 - ax1;
        float acy = ay1 + 0.5f * ah, acx = ax1 + 0.5f * aw;
        float ty = rlds[t * 4 + 0], tx = rlds[t * 4 + 1];
        float th = rlds[t * 4 + 2], tw = rlds[t * 4 + 3];
        float cy = acy + ty * ah, cx = acx + tx * aw;
        float h = ah * __expf(th), w = aw * __expf(tw);
        float y1 = fmaxf(cy - 0.5f * h, 0.f);
        float x1 = fmaxf(cx - 0.5f * w, 0.f);
        float y2 = fminf(cy + 0.5f * h, 1600.f);
        float x2 = fminf(cx + 0.5f * w, 2560.f);
        int gid = pix * 9 + t;
        props[gid * 4 + 0] = y1; props[gid * 4 + 1] = x1;
        props[gid * 4 + 2] = y2; props[gid * 4 + 3] = x2;
        float s = slds[t];
        obj[gid] = s;
        float hh = y2 - y1, ww = x2 - x1;
        scmask[gid] = (hh >= 16.f && ww >= 16.f) ? s : NEGV;
    }
}

// ---------------- top-k threshold via score bisection ----------------------
__global__ __launch_bounds__(1024) void topk_thresh(
    const float* __restrict__ obj, float* __restrict__ thr_out,
    int* __restrict__ counter, float* __restrict__ cscore, float* __restrict__ cbox)
{
    __shared__ float lohi[2];
    __shared__ int cnt[1024];
    const int t = threadIdx.x;
    if (t == 0) { lohi[0] = 0.f; lohi[1] = 1.f; counter[0] = 0; }
    for (int i = t; i < PRE_NMS; i += 1024) {
        cscore[i] = NEGV;
        cbox[i * 4 + 0] = 0.f; cbox[i * 4 + 1] = 0.f;
        cbox[i * 4 + 2] = 0.f; cbox[i * 4 + 3] = 0.f;
    }
    __syncthreads();
    for (int it = 0; it < 32; ++it) {
        float mid = 0.5f * (lohi[0] + lohi[1]);
        int c = 0;
        for (int i = t; i < NANCH; i += 1024) c += (obj[i] > mid) ? 1 : 0;
        cnt[t] = c;
        __syncthreads();
        for (int s = 512; s > 0; s >>= 1) {
            if (t < s) cnt[t] += cnt[t + s];
            __syncthreads();
        }
        if (t == 0) { if (cnt[0] > PRE_NMS) lohi[0] = mid; else lohi[1] = mid; }
        __syncthreads();
    }
    if (t == 0) thr_out[0] = lohi[1];   // count(> hi) <= PRE_NMS guaranteed
}

// ---------------- compact top-k survivors ----------------------------------
__global__ void compact_topk(
    const float* __restrict__ obj, const float* __restrict__ scmask,
    const float* __restrict__ props, const float* __restrict__ thr,
    int* __restrict__ counter, float* __restrict__ cscore, float* __restrict__ cbox)
{
    int i = blockIdx.x * blockDim.x + threadIdx.x;
    if (i >= NANCH) return;
    if (obj[i] > thr[0]) {
        int slot = atomicAdd(counter, 1);
        if (slot < PRE_NMS) {
            cscore[slot] = scmask[i];
            cbox[slot * 4 + 0] = props[i * 4 + 0];
            cbox[slot * 4 + 1] = props[i * 4 + 1];
            cbox[slot * 4 + 2] = props[i * 4 + 2];
            cbox[slot * 4 + 3] = props[i * 4 + 3];
        }
    }
}

// ---------------- greedy NMS (300 picks) -----------------------------------
__global__ __launch_bounds__(1024) void nms_kernel(
    float* __restrict__ cscore, const float* __restrict__ cbox,
    float* __restrict__ out_props)     // [POST_NMS][4]
{
    __shared__ float bval[1024];
    __shared__ int   bidx[1024];
    __shared__ float bbox[4];
    const int t = threadIdx.x;

    for (int it = 0; it < POST_NMS; ++it) {
        float best = -3.0e38f; int bi = 0x7fffffff;
        for (int i = t; i < PRE_NMS; i += 1024) {
            float s = cscore[i];
            if (s > best || (s == best && i < bi)) { best = s; bi = i; }
        }
        bval[t] = best; bidx[t] = bi;
        __syncthreads();
        for (int s = 512; s > 0; s >>= 1) {
            if (t < s) {
                if (bval[t + s] > bval[t] ||
                    (bval[t + s] == bval[t] && bidx[t + s] < bidx[t])) {
                    bval[t] = bval[t + s]; bidx[t] = bidx[t + s];
                }
            }
            __syncthreads();
        }
        if (t == 0) {
            int i = bidx[0];
            bool valid = (bval[0] > NEGV * 0.5f);
            float b0 = cbox[i * 4 + 0], b1 = cbox[i * 4 + 1];
            float b2 = cbox[i * 4 + 2], b3 = cbox[i * 4 + 3];
            bbox[0] = b0; bbox[1] = b1; bbox[2] = b2; bbox[3] = b3;
            out_props[it * 4 + 0] = valid ? b0 : 0.f;
            out_props[it * 4 + 1] = valid ? b1 : 0.f;
            out_props[it * 4 + 2] = valid ? b2 : 0.f;
            out_props[it * 4 + 3] = valid ? b3 : 0.f;
            cscore[i] = NEGV;
        }
        __syncthreads();
        float y1 = bbox[0], x1 = bbox[1], y2 = bbox[2], x2 = bbox[3];
        float a1 = fmaxf(y2 - y1, 0.f) * fmaxf(x2 - x1, 0.f);
        for (int i = t; i < PRE_NMS; i += 1024) {
            float by1 = cbox[i * 4 + 0], bx1 = cbox[i * 4 + 1];
            float by2 = cbox[i * 4 + 2], bx2 = cbox[i * 4 + 3];
            float yy1 = fmaxf(y1, by1), xx1 = fmaxf(x1, bx1);
            float yy2 = fminf(y2, by2), xx2 = fminf(x2, bx2);
            float inter = fmaxf(yy2 - yy1, 0.f) * fmaxf(xx2 - xx1, 0.f);
            float a2 = fmaxf(by2 - by1, 0.f) * fmaxf(bx2 - bx1, 0.f);
            float iou = inter / (a1 + a2 - inter + 1e-8f);
            if (iou > 0.7f) cscore[i] = NEGV;
        }
        __syncthreads();
    }
}

// ---------------------------------------------------------------------------
extern "C" void kernel_launch(void* const* d_in, const int* in_sizes, int n_in,
                              void* d_out, int out_size, void* d_ws, size_t ws_size,
                              hipStream_t stream)
{
    (void)in_sizes; (void)n_in; (void)out_size; (void)ws_size;
    const float* feat       = (const float*)d_in[1];
    const float* anchor_map = (const float*)d_in[2];
    const float* W1 = (const float*)d_in[4];
    const float* b1 = (const float*)d_in[5];
    const float* Wc = (const float*)d_in[6];
    const float* bc = (const float*)d_in[7];
    const float* Wr = (const float*)d_in[8];
    const float* br = (const float*)d_in[9];
    float* out = (float*)d_out;

    char* ws = (char*)d_ws;
    size_t off = 0;
    auto take = [&](size_t bytes) -> char* {
        char* p = ws + off;
        off = (off + bytes + 255) & ~(size_t)255;
        return p;
    };

    const size_t n_pad  = (size_t)PADH * PADW * CIN;                 // 8,460,288
    const size_t n_pack = (size_t)9 * 16 * 32 * 32 * 16;             // 2,359,296

    _Float16* featpad = (_Float16*)take(n_pad * sizeof(_Float16));
    _Float16* wpack   = (_Float16*)take(n_pack * sizeof(_Float16));
    float* y      = (float*)take((size_t)NPIX * CIN * sizeof(float));
    float* props  = (float*)take((size_t)NANCH * 4 * sizeof(float));
    float* obj    = (float*)take((size_t)NANCH * sizeof(float));
    float* scmask = (float*)take((size_t)NANCH * sizeof(float));
    float* cbox   = (float*)take((size_t)PRE_NMS * 4 * sizeof(float));
    float* cscore = (float*)take((size_t)PRE_NMS * sizeof(float));
    float* thr    = (float*)take(sizeof(float));
    int*   counter= (int*)take(sizeof(int));

    pad_feat_f16<<<(unsigned)(n_pad / 256), 256, 0, stream>>>(feat, featpad);
    pack_w1_f16<<<(unsigned)(n_pack / 256), 256, 0, stream>>>(W1, wpack);
    conv3x3_wmma<<<NPIX / 32, 256, 0, stream>>>(featpad, wpack, b1, y);
    heads_decode<<<NPIX, 64, 0, stream>>>(y, Wc, bc, Wr, br, anchor_map,
                                          out, out + NANCH, props, obj, scmask);
    topk_thresh<<<1, 1024, 0, stream>>>(obj, thr, counter, cscore, cbox);
    compact_topk<<<(NANCH + 255) / 256, 256, 0, stream>>>(obj, scmask, props, thr,
                                                          counter, cscore, cbox);
    nms_kernel<<<1, 1024, 0, stream>>>(cscore, cbox, out + (size_t)NANCH + NPIX * 36);
}
